// UltraHighFrequencyModel_37082747634596
// MI455X (gfx1250) — matrix operand order repaired and checked by reading, not compiled, not run on hardware
//
#include <hip/hip_runtime.h>
#include <cmath>

// ---------------------------------------------------------------------------
// Model constants (match reference)
// ---------------------------------------------------------------------------
static constexpr int H_  = 960;
static constexpr int H2_ = 480;
static constexpr int S_  = 512;
static constexpr int B_  = 2;
static constexpr int F_  = 64;
static constexpr int NQ_ = 9;
static constexpr int DFF_ = 512;

// ---------------------------------------------------------------------------
// WMMA types (gfx1250, wave32)
// ---------------------------------------------------------------------------
typedef __attribute__((ext_vector_type(16))) __bf16 v16bf;
typedef __attribute__((ext_vector_type(8)))  float  v8f;

__device__ __forceinline__ unsigned short f2bf(float f) {
  unsigned u = __builtin_bit_cast(unsigned, f);
  unsigned r = u + 0x7FFFu + ((u >> 16) & 1u);   // round-to-nearest-even
  return (unsigned short)(r >> 16);
}
__device__ __forceinline__ unsigned pack2bf(float a, float b) {
  return (unsigned)f2bf(a) | ((unsigned)f2bf(b) << 16);
}

__device__ __forceinline__ float actf(float x, int act) {
  switch (act) {
    case 1: return 0.5f * x * (1.0f + erff(x * 0.70710678118654752f)); // erf-GELU
    case 2: return fmaxf(x, 0.0f);                                     // ReLU
    case 3: return 1.0f / (1.0f + __expf(-x));                         // sigmoid
    default: return x;
  }
}

// A-fragment LDS layout for 16-bit WMMA A operand:
//   lane-half 0 holds K 0..15 ; lane-half 1 holds K 16..31
__device__ __forceinline__ int aPos(int m, int k) {
  return m * 32 + k;
}

#define TM 128
#define TN 64
#define TK 32

// ---------------------------------------------------------------------------
// Generic batched GEMM:  C[z] = act( scale * A[z]*B[z] + bias (+res) ) (+res)
//   A fp32 [M x K] (lda), B fp32 [K x N] (ldb) or [N x K] if transB,
//   bf16 WMMA 16x16x32, fp32 accumulate.
//   Batch offsets: off(z) = (z/nhz)*off1 + (z%nhz)*off2   (head-strided batching)
// WG = 256 threads = 8 waves (4x2); WG tile 128x64; wave tile 32x32 (2x2 WMMA).
// Two uniform-branch staging paths only:
//   fast   : full tile + full K-slab -> float2 loads, packed b32 LDS stores
//   general: clamp-address loads, result kept LIVE via mask-multiply so the
//            compiler cannot re-if-convert the load into an exec diamond
// ---------------------------------------------------------------------------
__global__ void __launch_bounds__(256)
gemm_wmma(const float* __restrict__ A, const float* __restrict__ Bm,
          const float* __restrict__ bias, const float* __restrict__ res,
          float* __restrict__ C,
          int M, int N, int K, int lda, int ldb, int ldc,
          long long oA1, long long oA2, long long oB1, long long oB2,
          long long oC1, long long oC2, int nhz, int transB,
          float scale, int act, int resPre)
{
  const int z = blockIdx.z;
  const long long zq = z / nhz, zr = z % nhz;
  A  += zq * oA1 + zr * oA2;
  Bm += zq * oB1 + zr * oB2;
  const long long zC = zq * oC1 + zr * oC2;
  C += zC;
  if (res) res += zC;

  __shared__ alignas(64) unsigned short sA[TM * TK];
  __shared__ alignas(64) unsigned short sB[TN * TK];

  const int tid  = threadIdx.x;
  const int bm   = blockIdx.y * TM;
  const int bn   = blockIdx.x * TN;
  const int wave = tid >> 5, lane = tid & 31;
  const int waveR = wave >> 1;
  const int waveC = wave & 1;
  const int halfl = lane >> 4;
  const int lr    = lane & 15;

  const bool fullA = (bm + TM <= M);
  const bool fullB = (bn + TN <= N);

  v8f acc[2][2] = {};

  for (int k0 = 0; k0 < K; k0 += TK) {
    const bool fullK = (k0 + TK <= K);

    // ------------------------- stage A (128 x 32) -------------------------
    if (fullA && fullK) {
      for (int i = tid; i < TM * TK / 2; i += 256) {
        int m = i >> 4;
        int kc = (i & 15) * 2;
        float2 f = *(const float2*)&A[(long long)(bm + m) * lda + k0 + kc];
        *(unsigned*)&sA[aPos(m, kc)] = pack2bf(f.x, f.y);
      }
      if (k0 + TK < K) {
        __builtin_prefetch(&A[(long long)(bm + (tid >> 1)) * lda + k0 + TK], 0, 3);
      }
    } else {
      // clamp + mask-multiply: unconditional load, flat select, no exec waves
      for (int i = tid; i < TM * TK; i += 256) {
        int m = i >> 5, k = i & 31;
        int gm = bm + m, gk = k0 + k;
        int gmc = gm < M ? gm : M - 1;
        int gkc = gk < K ? gk : K - 1;
        float msk = (gm < M && gk < K) ? 1.0f : 0.0f;
        float f = A[(long long)gmc * lda + gkc] * msk;
        sA[aPos(m, k)] = f2bf(f);
      }
    }

    // ------------------- stage B (32 x 64) as [n][k] ----------------------
    if (transB) {
      if (fullB && fullK) {
        for (int i = tid; i < TN * TK / 2; i += 256) {
          int n = i >> 4;
          int kc = (i & 15) * 2;
          float2 f = *(const float2*)&Bm[(long long)(bn + n) * ldb + k0 + kc];
          *(unsigned*)&sB[n * 32 + kc] = pack2bf(f.x, f.y);
        }
      } else {
        for (int i = tid; i < TN * TK; i += 256) {
          int n = i >> 5, k = i & 31;
          int gn = bn + n, gk = k0 + k;
          int gnc = gn < N ? gn : N - 1;
          int gkc = gk < K ? gk : K - 1;
          float msk = (gn < N && gk < K) ? 1.0f : 0.0f;
          float f = Bm[(long long)gnc * ldb + gkc] * msk;
          sB[n * 32 + k] = f2bf(f);
        }
      }
    } else {
      if (fullB && fullK) {
        for (int i = tid; i < TN * TK / 2; i += 256) {
          int k = i >> 5;
          int nc = (i & 31) * 2;
          float2 f = *(const float2*)&Bm[(long long)(k0 + k) * ldb + bn + nc];
          sB[nc * 32 + k]       = f2bf(f.x);
          sB[(nc + 1) * 32 + k] = f2bf(f.y);
        }
      } else {
        for (int i = tid; i < TN * TK; i += 256) {
          int n = i >> 5, k = i & 31;
          int gn = bn + n, gk = k0 + k;
          int gnc = gn < N ? gn : N - 1;
          int gkc = gk < K ? gk : K - 1;
          float msk = (gn < N && gk < K) ? 1.0f : 0.0f;
          float f = Bm[(long long)gkc * ldb + gnc] * msk;
          sB[n * 32 + k] = f2bf(f);
        }
      }
    }
    __syncthreads();

#pragma unroll
    for (int mi = 0; mi < 2; ++mi) {
      int arow = waveR * 32 + mi * 16 + lr;
      v16bf av = *(const v16bf*)&sA[arow * 32 + halfl * 16];
#pragma unroll
      for (int ni = 0; ni < 2; ++ni) {
        int bcol = waveC * 32 + ni * 16 + lr;
        v16bf bv = *(const v16bf*)&sB[bcol * 32 + halfl * 16];
        acc[mi][ni] = __builtin_amdgcn_wmma_f32_16x16x32_bf16(
            false, av, false, bv, (short)0, acc[mi][ni], false, false);
      }
    }
    __syncthreads();
  }

  // ---- epilogue: C/D layout -> row M = r + (lane/16)*8, col N = lane%16 ----
  if (fullA && fullB) {
#pragma unroll
    for (int mi = 0; mi < 2; ++mi)
#pragma unroll
      for (int ni = 0; ni < 2; ++ni)
#pragma unroll
        for (int r = 0; r < 8; ++r) {
          int gm = bm + waveR * 32 + mi * 16 + halfl * 8 + r;
          int gn = bn + waveC * 32 + ni * 16 + lr;
          float v = acc[mi][ni][r] * scale;
          if (bias) v += bias[gn];
          long long ci = (long long)gm * ldc + gn;
          if (resPre && res) v += res[ci];
          v = actf(v, act);
          if (!resPre && res) v += res[ci];
          C[ci] = v;
        }
  } else {
#pragma unroll
    for (int mi = 0; mi < 2; ++mi)
#pragma unroll
      for (int ni = 0; ni < 2; ++ni)
#pragma unroll
        for (int r = 0; r < 8; ++r) {
          int gm = bm + waveR * 32 + mi * 16 + halfl * 8 + r;
          int gn = bn + waveC * 32 + ni * 16 + lr;
          if (gm < M && gn < N) {
            float v = acc[mi][ni][r] * scale;
            if (bias) v += bias[gn];
            long long ci = (long long)gm * ldc + gn;
            if (resPre && res) v += res[ci];
            v = actf(v, act);
            if (!resPre && res) v += res[ci];
            C[ci] = v;
          }
        }
  }
}

// ---------------------------------------------------------------------------
// Conv1d as im2col WMMA GEMM (CIN/SN compile-time -> div/mod by constants).
//   y[b,s,o] = bias[o] + sum_{kk,ci} x[b, s+kk-pad, ci] * w[o, ci, kk]
//   M = B*S rows (b,s), N = Cout, K = Cin*ksz with col gk = kk*Cin + ci.
//   K multiple of 32, M multiple of 128, N multiple of 64 -> branch-free
//   staging; padding handled with clamp + mask-multiply (flat select).
// ---------------------------------------------------------------------------
template <int CIN, int SN>
__global__ void __launch_bounds__(256)
conv_gemm_wmma(const float* __restrict__ X, const float* __restrict__ W,
               const float* __restrict__ bias, float* __restrict__ Y,
               int M, int N, int K, int ksz, int pad, int chanOff, int ldc)
{
  __shared__ alignas(64) unsigned short sA[TM * TK];
  __shared__ alignas(64) unsigned short sB[TN * TK];

  const int tid  = threadIdx.x;
  const int bm   = blockIdx.y * TM;
  const int bn   = blockIdx.x * TN;
  const int wave = tid >> 5, lane = tid & 31;
  const int waveR = wave >> 1;
  const int waveC = wave & 1;
  const int halfl = lane >> 4;
  const int lr    = lane & 15;

  v8f acc[2][2] = {};

  for (int k0 = 0; k0 < K; k0 += TK) {
    // ---- A: im2col gather, clamp + mask-multiply padding ----
    for (int i = tid; i < TM * TK; i += 256) {
      int m = i >> 5, k = i & 31;
      int gm = bm + m, gk = k0 + k;
      int kk = gk / CIN, ci = gk % CIN;          // compile-time divisor
      int b  = gm / SN,  s  = gm % SN;
      int sp = s + kk - pad;
      int spc = sp < 0 ? 0 : (sp >= SN ? SN - 1 : sp);
      float msk = (sp >= 0 && sp < SN) ? 1.0f : 0.0f;
      float f = X[((long long)b * SN + spc) * CIN + ci] * msk;
      sA[aPos(m, k)] = f2bf(f);
    }
    // ---- B: permuted weight gather  B(gk, n) = w[n][ci][kk] ----
    for (int i = tid; i < TN * TK; i += 256) {
      int n = i >> 5, k = i & 31;
      int gk = k0 + k;
      int kk = gk / CIN, ci = gk % CIN;
      float f = W[((long long)(bn + n) * CIN + ci) * ksz + kk];
      sB[n * 32 + k] = f2bf(f);
    }
    __syncthreads();

#pragma unroll
    for (int mi = 0; mi < 2; ++mi) {
      int arow = waveR * 32 + mi * 16 + lr;
      v16bf av = *(const v16bf*)&sA[arow * 32 + halfl * 16];
#pragma unroll
      for (int ni = 0; ni < 2; ++ni) {
        int bcol = waveC * 32 + ni * 16 + lr;
        v16bf bv = *(const v16bf*)&sB[bcol * 32 + halfl * 16];
        acc[mi][ni] = __builtin_amdgcn_wmma_f32_16x16x32_bf16(
            false, av, false, bv, (short)0, acc[mi][ni], false, false);
      }
    }
    __syncthreads();
  }

#pragma unroll
  for (int mi = 0; mi < 2; ++mi)
#pragma unroll
    for (int ni = 0; ni < 2; ++ni)
#pragma unroll
      for (int r = 0; r < 8; ++r) {
        int gm = bm + waveR * 32 + mi * 16 + halfl * 8 + r;
        int gn = bn + waveC * 32 + ni * 16 + lr;
        float v = acc[mi][ni][r] + bias[gn];
        Y[(long long)gm * ldc + chanOff + gn] = v;
      }
}

// ---------------------------------------------------------------------------
// Row softmax with optional per-head temperature
// ---------------------------------------------------------------------------
__global__ void __launch_bounds__(256)
softmax_rows(float* __restrict__ p, const float* __restrict__ temp,
             int nh, int rowlen, float scale)
{
  long long row = blockIdx.x;
  float* pr = p + row * rowlen;
  int head = (int)((row / rowlen) % nh);
  float mul = scale * (temp ? temp[head] : 1.0f);

  __shared__ float red[256];
  int tid = threadIdx.x;

  float mx = -1e30f;
  for (int i = tid; i < rowlen; i += 256) mx = fmaxf(mx, pr[i] * mul);
  red[tid] = mx; __syncthreads();
  for (int o = 128; o > 0; o >>= 1) {
    if (tid < o) red[tid] = fmaxf(red[tid], red[tid + o]);
    __syncthreads();
  }
  mx = red[0]; __syncthreads();

  float sum = 0.0f;
  for (int i = tid; i < rowlen; i += 256) {
    float e = __expf(pr[i] * mul - mx);
    pr[i] = e; sum += e;
  }
  red[tid] = sum; __syncthreads();
  for (int o = 128; o > 0; o >>= 1) {
    if (tid < o) red[tid] += red[tid + o];
    __syncthreads();
  }
  float inv = 1.0f / red[0];
  for (int i = tid; i < rowlen; i += 256) pr[i] *= inv;
}

// ---------------------------------------------------------------------------
// LayerNorm: y = act( ((x+resPre)-mean)/sqrt(var+eps)*g + b ) + resPost
// ---------------------------------------------------------------------------
__global__ void __launch_bounds__(256)
layernorm_k(const float* __restrict__ x, const float* __restrict__ resPre,
            const float* __restrict__ g, const float* __restrict__ b,
            float* __restrict__ y, const float* __restrict__ resPost,
            int Hn, int act)
{
  long long row = blockIdx.x;
  const float* xr = x + row * Hn;
  const float* rp = resPre  ? resPre  + row * Hn : nullptr;
  const float* rq = resPost ? resPost + row * Hn : nullptr;
  float* yr = y + row * Hn;
  __shared__ float red[256];
  int tid = threadIdx.x;

  float s = 0.0f;
  for (int i = tid; i < Hn; i += 256) {
    float v = xr[i]; if (rp) v += rp[i];
    s += v;
  }
  red[tid] = s; __syncthreads();
  for (int o = 128; o > 0; o >>= 1) { if (tid < o) red[tid] += red[tid + o]; __syncthreads(); }
  float mean = red[0] / (float)Hn; __syncthreads();

  float vs = 0.0f;
  for (int i = tid; i < Hn; i += 256) {
    float v = xr[i]; if (rp) v += rp[i];
    float d = v - mean; vs += d * d;
  }
  red[tid] = vs; __syncthreads();
  for (int o = 128; o > 0; o >>= 1) { if (tid < o) red[tid] += red[tid + o]; __syncthreads(); }
  float rstd = rsqrtf(red[0] / (float)Hn + 1e-5f);

  for (int i = tid; i < Hn; i += 256) {
    float v = xr[i]; if (rp) v += rp[i];
    float o = actf((v - mean) * rstd * g[i] + b[i], act);
    if (rq) o += rq[i];
    yr[i] = o;
  }
}

// ---------------------------------------------------------------------------
// Broadcast freq embedding into h[:, :, H2:H]
// ---------------------------------------------------------------------------
__global__ void __launch_bounds__(256)
freqcopy_k(const float* __restrict__ freq, float* __restrict__ h,
           int Bn, int Sn, int Hn, int H2n)
{
  int idx = blockIdx.x * blockDim.x + threadIdx.x;
  int total = Bn * Sn * H2n;
  if (idx >= total) return;
  int c = idx % H2n;
  int s = (idx / H2n) % Sn;
  int b = idx / (H2n * Sn);
  h[((long long)b * Sn + s) * Hn + H2n + c] = freq[(long long)s * H2n + c];
}

// ---------------------------------------------------------------------------
// GRN combine: y = x + g*p1 + (1-g)*p2
// ---------------------------------------------------------------------------
__global__ void __launch_bounds__(256)
grn_combine_k(const float* __restrict__ x, const float* __restrict__ g,
              const float* __restrict__ p1, const float* __restrict__ p2,
              float* __restrict__ y, int n)
{
  int i = blockIdx.x * blockDim.x + threadIdx.x;
  if (i < n) {
    float gg = g[i];
    y[i] = x[i] + gg * p1[i] + (1.0f - gg) * p2[i];
  }
}

// ---------------------------------------------------------------------------
// Attention pooling: aw = softmax(logits / pool_t over S); pooled = sum_s h*aw
// ---------------------------------------------------------------------------
__global__ void __launch_bounds__(256)
pool_k(const float* __restrict__ h, const float* __restrict__ logits,
       const float* __restrict__ poolT, float* __restrict__ pooled,
       int Sn, int Hn)
{
  int b = blockIdx.x;
  __shared__ float w[512];
  __shared__ float red[256];
  int tid = threadIdx.x;
  float invT = 1.0f / poolT[0];
  const float* lg = logits + (long long)b * Sn;

  float mx = -1e30f;
  for (int s = tid; s < Sn; s += 256) mx = fmaxf(mx, lg[s] * invT);
  red[tid] = mx; __syncthreads();
  for (int o = 128; o > 0; o >>= 1) { if (tid < o) red[tid] = fmaxf(red[tid], red[tid + o]); __syncthreads(); }
  mx = red[0]; __syncthreads();

  float sum = 0.0f;
  for (int s = tid; s < Sn; s += 256) { float e = __expf(lg[s] * invT - mx); w[s] = e; sum += e; }
  red[tid] = sum; __syncthreads();
  for (int o = 128; o > 0; o >>= 1) { if (tid < o) red[tid] += red[tid + o]; __syncthreads(); }
  float inv = 1.0f / red[0];
  __syncthreads();

  for (int c = tid; c < Hn; c += 256) {
    float acc = 0.0f;
    for (int s = 0; s < Sn; ++s) acc += h[((long long)b * Sn + s) * Hn + c] * w[s];
    pooled[(long long)b * Hn + c] = acc * inv;
  }
}

// ===========================================================================
// Host-side orchestration
// ===========================================================================
struct AttnP { const float *qw,*qb,*kw,*kb,*vw,*vb,*ow,*ob,*temp; };

static void gemm(hipStream_t st, const float* A, const float* Bm, const float* bias,
                 const float* res, float* C, int M, int N, int K,
                 int lda, int ldb, int ldc, int nz,
                 long long oA1, long long oA2, long long oB1, long long oB2,
                 long long oC1, long long oC2, int nhz, int transB,
                 float scale, int act, int resPre)
{
  dim3 grid((unsigned)((N + TN - 1) / TN), (unsigned)((M + TM - 1) / TM), (unsigned)nz);
  gemm_wmma<<<grid, 256, 0, st>>>(A, Bm, bias, res, C, M, N, K, lda, ldb, ldc,
                                  oA1, oA2, oB1, oB2, oC1, oC2, nhz, transB,
                                  scale, act, resPre);
}

static void run_mha(hipStream_t st, const float* hin, const AttnP& p, int nh,
                    float* q, float* k, float* v, float* ctx, float* scores,
                    float* out, const float* resid, bool useTemp)
{
  const int Mr = B_ * S_;
  gemm(st, hin, p.qw, p.qb, nullptr, q, Mr, H_, H_, H_, H_, H_, 1,0,0,0,0,0,0,1,0,1.0f,0,0);
  gemm(st, hin, p.kw, p.kb, nullptr, k, Mr, H_, H_, H_, H_, H_, 1,0,0,0,0,0,0,1,0,1.0f,0,0);
  gemm(st, hin, p.vw, p.vb, nullptr, v, Mr, H_, H_, H_, H_, H_, 1,0,0,0,0,0,0,1,0,1.0f,0,0);

  const int hd = H_ / nh;
  const float inv = 1.0f / sqrtf((float)hd);
  const int CH = 8;                       // head chunk (caps score memory)
  for (int c = 0; c < nh; c += CH) {
    int nc = nh - c; if (nc > CH) nc = CH;
    int nz = B_ * nc;
    gemm(st, q + c * hd, k + c * hd, nullptr, nullptr, scores,
         S_, S_, hd, H_, H_, S_, nz,
         (long long)S_ * H_, (long long)hd,
         (long long)S_ * H_, (long long)hd,
         (long long)nc * S_ * S_, (long long)S_ * S_,
         nc, /*transB=*/1, 1.0f, 0, 0);
    softmax_rows<<<dim3((unsigned)(nz * S_)), 256, 0, st>>>(
        scores, useTemp ? p.temp + c : nullptr, nc, S_, inv);
    gemm(st, scores, v + c * hd, nullptr, nullptr, ctx + c * hd,
         S_, hd, S_, S_, H_, H_, nz,
         (long long)nc * S_ * S_, (long long)S_ * S_,
         (long long)S_ * H_, (long long)hd,
         (long long)S_ * H_, (long long)hd,
         nc, 0, 1.0f, 0, 0);
  }
  gemm(st, ctx, p.ow, p.ob, resid, out, Mr, H_, H_, H_, H_, H_, 1,0,0,0,0,0,0,1,0,1.0f,0,0);
}

extern "C" void kernel_launch(void* const* d_in, const int* in_sizes, int n_in,
                              void* d_out, int out_size, void* d_ws, size_t ws_size,
                              hipStream_t stream)
{
  (void)in_sizes; (void)n_in; (void)out_size; (void)ws_size;
  int ii = 0;
  auto nx = [&]() { return (const float*)d_in[ii++]; };

  const float* x      = nx();
  const float* feat_w = nx();
  const float* feat_b = nx();
  const float* freq   = nx();

  AttnP hier[3];
  for (int i = 0; i < 3; ++i) {
    hier[i].qw = nx(); hier[i].qb = nx(); hier[i].kw = nx(); hier[i].kb = nx();
    hier[i].vw = nx(); hier[i].vb = nx(); hier[i].ow = nx(); hier[i].ob = nx();
    hier[i].temp = nx();
  }
  struct FlowP { const float *w1,*b1,*w2,*b2; } flow[3];
  for (int i = 0; i < 3; ++i) { flow[i].w1 = nx(); flow[i].b1 = nx(); flow[i].w2 = nx(); flow[i].b2 = nx(); }
  struct ConvP { const float* kw[5]; const float* kb[5]; const float *g,*b; } conv[4];
  for (int i = 0; i < 4; ++i) {
    for (int j = 0; j < 5; ++j) conv[i].kw[j] = nx();
    for (int j = 0; j < 5; ++j) conv[i].kb[j] = nx();
    conv[i].g = nx(); conv[i].b = nx();
  }
  struct LayerP { AttnP a; const float *n1g,*n1b,*n2g,*n2b,*fw1,*fb1,*fw2,*fb2,*qhw,*qhb; } layers[8];
  for (int i = 0; i < 8; ++i) {
    LayerP& L = layers[i];
    L.a.qw = nx(); L.a.qb = nx(); L.a.kw = nx(); L.a.kb = nx();
    L.a.vw = nx(); L.a.vb = nx(); L.a.ow = nx(); L.a.ob = nx(); L.a.temp = nx();
    L.n1g = nx(); L.n1b = nx(); L.n2g = nx(); L.n2b = nx();
    L.fw1 = nx(); L.fb1 = nx(); L.fw2 = nx(); L.fb2 = nx();
    L.qhw = nx(); L.qhb = nx();
  }
  struct RegP { AttnP a; const float *n1g,*n1b,*n2g,*n2b,*fw1,*fb1,*fw2,*fb2; } regime[2];
  for (int i = 0; i < 2; ++i) {
    RegP& R = regime[i];
    R.a.qw = nx(); R.a.qb = nx(); R.a.kw = nx(); R.a.kb = nx();
    R.a.vw = nx(); R.a.vb = nx(); R.a.ow = nx(); R.a.ob = nx(); R.a.temp = nx();
    R.n1g = nx(); R.n1b = nx(); R.n2g = nx(); R.n2b = nx();
    R.fw1 = nx(); R.fb1 = nx(); R.fw2 = nx(); R.fb2 = nx();
  }
  const float* pool_w = nx();
  const float* pool_b = nx();
  const float* pool_t = nx();
  struct HeadP { const float *lng,*lnb,*w1,*b1,*w2,*b2,*gw,*gb,*l1w,*l1b,*l2w,*l2b; } heads[2];
  for (int i = 0; i < 2; ++i) {
    HeadP& Hh = heads[i];
    Hh.lng = nx(); Hh.lnb = nx(); Hh.w1 = nx(); Hh.b1 = nx();
    Hh.w2 = nx(); Hh.b2 = nx(); Hh.gw = nx(); Hh.gb = nx();
    Hh.l1w = nx(); Hh.l1b = nx(); Hh.l2w = nx(); Hh.l2b = nx();
  }

  // ---- workspace carve (floats) ----
  float* ws = (float*)d_ws;
  auto carve = [&](size_t n) { float* p = ws; ws += n; return p; };
  const size_t tok = (size_t)B_ * S_ * H_;
  float* hA     = carve(tok);
  float* hB     = carve(tok);
  float* q      = carve(tok);
  float* k      = carve(tok);
  float* v      = carve(tok);
  float* ctx    = carve(tok);
  float* ffbuf  = carve((size_t)B_ * S_ * 4 * H_);
  float* scores = carve((size_t)B_ * 8 * S_ * S_);
  float* plog   = carve((size_t)B_ * S_);
  float* pooled = carve((size_t)B_ * H_);
  float* hg     = carve((size_t)B_ * H_);
  float* p1b    = carve((size_t)B_ * H_);
  float* p2b    = carve((size_t)B_ * H_);
  float* g1     = carve((size_t)B_ * H_);
  float* g2     = carve((size_t)B_ * H_);
  float* zb     = carve((size_t)B_ * H_);

  float* outp = (float*)d_out;   // [policy(6) | value(18) | qp(9216)]
  const int Mr = B_ * S_;

  // ---- 1. feature projection + freq broadcast ----
  gemm(stream, x, feat_w, feat_b, nullptr, hA, Mr, H2_, F_, F_, H2_, H_,
       1,0,0,0,0,0,0,1,0,1.0f,0,0);
  {
    int total = B_ * S_ * H2_;
    freqcopy_k<<<dim3((unsigned)((total + 255) / 256)), 256, 0, stream>>>(
        freq, hA, B_, S_, H_, H2_);
  }

  // ---- 2. hierarchical adaptive attention (residual), nh=16 ----
  for (int i = 0; i < 3; ++i) {
    run_mha(stream, hA, hier[i], 16, q, k, v, ctx, scores, hB, hA, true);
    float* t = hA; hA = hB; hB = t;
  }

  // ---- 3. neural flow MLPs (residual) ----
  for (int i = 0; i < 3; ++i) {
    gemm(stream, hA, flow[i].w1, flow[i].b1, nullptr, ffbuf, Mr, H_, H_,
         H_, H_, H_, 1,0,0,0,0,0,0,1,0,1.0f,/*gelu*/1,0);
    gemm(stream, ffbuf, flow[i].w2, flow[i].b2, hA, hB, Mr, H_, H_,
         H_, H_, H_, 1,0,0,0,0,0,0,1,0,1.0f,0,0);
    float* t = hA; hA = hB; hB = t;
  }

  // ---- 4. multi-scale conv blocks as im2col WMMA GEMMs (residual) ----
  static const int KS_[5] = {3, 7, 15, 31, 63};
  for (int i = 0; i < 4; ++i) {
    for (int j = 0; j < 5; ++j) {
      int Cout = H_ / 5, ksz = KS_[j], pad = ksz / 2;
      dim3 grid((unsigned)((Cout + TN - 1) / TN), (unsigned)((Mr + TM - 1) / TM), 1);
      conv_gemm_wmma<H_, S_><<<grid, 256, 0, stream>>>(
          hA, conv[i].kw[j], conv[i].kb[j], ffbuf,
          Mr, Cout, H_ * ksz, ksz, pad, j * Cout, H_);
    }
    layernorm_k<<<dim3((unsigned)Mr), 256, 0, stream>>>(
        ffbuf, nullptr, conv[i].g, conv[i].b, hB, /*resPost=*/hA, H_, /*gelu*/1);
    float* t = hA; hA = hB; hB = t;
  }

  // ---- 5. QuantileTransformer layers, nh=32 ----
  for (int i = 0; i < 8; ++i) {
    const LayerP& L = layers[i];
    run_mha(stream, hA, L.a, 32, q, k, v, ctx, scores, hB, hA, true);
    layernorm_k<<<dim3((unsigned)Mr), 256, 0, stream>>>(
        hB, nullptr, L.n1g, L.n1b, hA, nullptr, H_, 0);
    gemm(stream, hA, L.fw1, L.fb1, nullptr, ffbuf, Mr, 4 * H_, H_,
         H_, 4 * H_, 4 * H_, 1,0,0,0,0,0,0,1,0,1.0f,/*gelu*/1,0);
    gemm(stream, ffbuf, L.fw2, L.fb2, hA, hB, Mr, H_, 4 * H_,
         4 * H_, H_, H_, 1,0,0,0,0,0,0,1,0,1.0f,0,0);
    layernorm_k<<<dim3((unsigned)Mr), 256, 0, stream>>>(
        hB, nullptr, L.n2g, L.n2b, hA, nullptr, H_, 0);
    if (i == 7) {
      gemm(stream, hA, L.qhw, L.qhb, nullptr, outp + 24, Mr, NQ_, H_,
           H_, NQ_, NQ_, 1,0,0,0,0,0,0,1,0,1.0f,0,0);
    }
  }

  // ---- 6. regime blocks ----
  for (int i = 0; i < 2; ++i) {
    const RegP& R = regime[i];
    run_mha(stream, hA, R.a, 32, q, k, v, ctx, scores, hB, hA, false);
    layernorm_k<<<dim3((unsigned)Mr), 256, 0, stream>>>(
        hB, nullptr, R.n1g, R.n1b, hA, nullptr, H_, 0);
    gemm(stream, hA, R.fw1, R.fb1, nullptr, ffbuf, Mr, DFF_, H_,
         H_, DFF_, DFF_, 1,0,0,0,0,0,0,1,0,1.0f,/*relu*/2,0);
    gemm(stream, ffbuf, R.fw2, R.fb2, hA, hB, Mr, H_, DFF_,
         DFF_, H_, H_, 1,0,0,0,0,0,0,1,0,1.0f,0,0);
    layernorm_k<<<dim3((unsigned)Mr), 256, 0, stream>>>(
        hB, nullptr, R.n2g, R.n2b, hA, nullptr, H_, 0);
  }

  // ---- 7. attention pooling ----
  gemm(stream, hA, pool_w, pool_b, nullptr, plog, Mr, 1, H_,
       H_, 1, 1, 1,0,0,0,0,0,0,1,0,1.0f,0,0);
  pool_k<<<dim3((unsigned)B_), 256, 0, stream>>>(hA, plog, pool_t, pooled, S_, H_);

  // ---- 8. GRN heads ----
  for (int hidx = 0; hidx < 2; ++hidx) {
    const HeadP& Hh = heads[hidx];
    int Nout = (hidx == 0) ? 3 : NQ_;
    float* dst = (hidx == 0) ? outp : outp + 6;
    layernorm_k<<<dim3((unsigned)B_), 256, 0, stream>>>(
        pooled, nullptr, Hh.lng, Hh.lnb, hg, nullptr, H_, 0);
    gemm(stream, hg, Hh.w1, Hh.b1, nullptr, p1b, B_, H_, H_, H_, H_, H_,
         1,0,0,0,0,0,0,1,0,1.0f,0,0);
    gemm(stream, hg, Hh.w2, Hh.b2, nullptr, p2b, B_, H_, H_, H_, H_, H_,
         1,0,0,0,0,0,0,1,0,1.0f,0,0);
    gemm(stream, p1b, Hh.gw, nullptr, nullptr, g1, B_, H_, H_, H_, H_, H_,
         1,0,0,0,0,0,0,1,0,1.0f,0,0);
    gemm(stream, p2b, Hh.gw + (long long)H_ * H_, nullptr, g1, g2,
         B_, H_, H_, H_, H_, H_, 1,0,0,0,0,0,0,1,0,1.0f,0,0);
    gemm(stream, hg, Hh.gw + 2LL * H_ * H_, Hh.gb, g2, g1,
         B_, H_, H_, H_, H_, H_, 1,0,0,0,0,0,0,1,0,1.0f,/*sigmoid*/3,/*resPre*/1);
    grn_combine_k<<<dim3((unsigned)((B_ * H_ + 255) / 256)), 256, 0, stream>>>(
        pooled, g1, p1b, p2b, zb, B_ * H_);
    gemm(stream, zb, Hh.l1w, Hh.l1b, nullptr, hg, B_, H_, H_, H_, H_, H_,
         1,0,0,0,0,0,0,1,0,1.0f,/*gelu*/1,0);
    gemm(stream, hg, Hh.l2w, Hh.l2b, nullptr, dst, B_, Nout, H_,
         H_, Nout, Nout, 1,0,0,0,0,0,0,1,0,1.0f,0,0);
  }
}
